// VQVAE_38843684225128
// MI455X (gfx1250) — matrix operand order repaired and checked by reading, not compiled, not run on hardware
//
#include <hip/hip_runtime.h>
#include <hip/hip_bf16.h>
#include <stdint.h>

typedef __attribute__((ext_vector_type(16))) __bf16 v16bf;
typedef __attribute__((ext_vector_type(8)))  float  v8f;

__device__ __forceinline__ unsigned short f2bf(float f) {
  unsigned u = __float_as_uint(f);
  unsigned rnd = 0x7FFFu + ((u >> 16) & 1u);       // round-to-nearest-even
  return (unsigned short)((u + rnd) >> 16);
}
__device__ __forceinline__ float bf2f(unsigned short h) {
  return __uint_as_float((unsigned)h << 16);
}
__device__ __forceinline__ int iclamp(int v, int lo, int hi) {
  return v < lo ? lo : (v > hi ? hi : v);
}

// ---------------------------------------------------------------------------
// Weight packing into WMMA A-fragment order (bf16).  Cold path, runs once.
// K-order is TAP-MAJOR: kk = tap*CI + ci  (so B fragments are NHWC-contiguous)
// A-frag layout (16x32 bf16, wave32): lane L = (m&15) + 16*half,
//   element e -> K = 16*half + e.
// Linear layout: dst[((mtile*Ktiles + kt)*32 + lane)*16 + e]
// ---------------------------------------------------------------------------
struct PackP {
  const float* w; unsigned short* dst;
  int mode;           // 0: conv (Co,Ci,KH,KW)  1: flip+transpose 3x3 (in,out,3,3)
                      // 2: dec2 phase (in,32,4,4)  3: plain MxK matrix
  int Co, Ci, KH, KW, Kpad, Kreal, Ktiles, a, b2;
};

__global__ void pack_a(PackP p) {
  int i = blockIdx.x * 256 + threadIdx.x;
  int total = p.Co * p.Kpad;
  if (i >= total) return;
  int m = i / p.Kpad, kk = i % p.Kpad;
  float val = 0.0f;
  if (kk < p.Kreal) {
    int ci = kk % p.Ci;                   // tap-major K order
    int r  = kk / p.Ci;                   // tap index
    if (p.mode == 0) {
      int kh = r / p.KW, kw = r % p.KW;
      val = p.w[((m * p.Ci + ci) * p.KH + kh) * p.KW + kw];
    } else if (p.mode == 1) {             // convT stride1: w[i][o][2-kh][2-kw]
      int kh = r / 3, kw = r % 3;
      val = p.w[((ci * p.Co + m) * 3 + (2 - kh)) * 3 + (2 - kw)];
    } else if (p.mode == 2) {             // convT stride2 phase taps
      int ti = r >> 1, tj = r & 1;
      int kh = p.a  ? (2 * ti) : (1 + 2 * ti);
      int kw = p.b2 ? (2 * tj) : (1 + 2 * tj);
      val = p.w[((ci * p.Co + m) * 4 + kh) * 4 + kw];
    } else {                              // plain matrix (codebook), K = ci
      val = p.w[m * p.Ci + kk];
    }
  }
  int mtile = m >> 4, lm = m & 15, kt = kk >> 5, kl = kk & 31;
  int half = kl >> 4;
  int e = kl & 15;
  p.dst[(size_t)((mtile * p.Ktiles + kt) * 32 + lm + 16 * half) * 16 + e] = f2bf(val);
}

// ---------------------------------------------------------------------------
// Fully specialized implicit-GEMM conv via V_WMMA_F32_16X16X32_BF16.
// The whole packed-A matrix (shared by all 4 waves of the block) is staged in
// LDS once per workgroup; steady-state A-fragment reads are ds_load pairs.
// Activations flow in bf16 NHWC: each 16-element B-fragment half = one tap,
// 16 consecutive channels = 2x global_load_b128.  One B gather feeds MT WMMAs.
// MODE 0: conv (ih = oh*STRIDE - PAD + kh); CI==1 uses scalar f32 gather (x).
// MODE 1: convT stride-2 parity phase (ih = ohb - ti, 2x2 taps).
// EP 0: relu->bf16 NHWC  EP 1: ->bf16 NHWC  EP 2: VQ argmin packed atomicMin
// ---------------------------------------------------------------------------
template<int MODE, int EP, int MT, int GROUPS, int CI, int H, int W,
         int WHS, int WS, int KH, int KW, int STRIDE, int PAD, int KTILES,
         int OH, int OW, int CO, int AA, int BB>
__global__ __launch_bounds__(128) void wmma_gemm(
    const unsigned short* __restrict__ A, const void* __restrict__ inv,
    const float* __restrict__ bias, void* __restrict__ outv,
    unsigned long long* __restrict__ vqmin, const float* __restrict__ cnorm) {
  constexpr bool SCALARB = (MODE == 0 && CI == 1);          // conv1 (reads f32)
  constexpr bool NOCHK   = (MODE == 0 && KH == 1 && PAD == 0);  // VQ 1x1
  constexpr int  AELEMS  = GROUPS * MT * KTILES * 512;      // whole packed A
  constexpr int  A4      = AELEMS / 8;                      // uint4 count

  __shared__ unsigned short Alds[AELEMS];

  const float* inf = (const float*)inv;
  const unsigned short* inb = (const unsigned short*)inv;

  const int tid  = threadIdx.x;
  const int lane = tid & 31;
  const int wave = tid >> 5;
  const int tile = blockIdx.x * 4 + wave;
  const int mgrp = (GROUPS == 1) ? 0 : (tile & (GROUPS - 1));   // GROUPS pow2
  const int ntile = (GROUPS == 1) ? tile : (tile >> (GROUPS == 4 ? 2 : 1));
  const int g = lane >> 4;                    // lane half-group
  const int col = ntile * 16 + (lane & 15);   // GEMM column (b,oh,ow)
  const int b   = col >> WHS;
  const int rem = col & ((1 << WHS) - 1);
  const int oh  = rem >> WS;
  const int ow  = rem & ((1 << WS) - 1);

  // ---- cooperative A staging: global b128 -> LDS (A4 is a multiple of 128)
  #pragma unroll 1
  for (int i = tid; i < A4; i += 128)
    ((uint4*)Alds)[i] = ((const uint4*)A)[i];
  __syncthreads();

  int ohb, owb;
  if (MODE == 0) { ohb = oh * STRIDE - PAD; owb = ow * STRIDE - PAD; }
  else           { ohb = AA ? oh + 1 : oh;  owb = BB ? ow + 1 : ow;  }

  const unsigned short* ap = Alds + (mgrp * MT * KTILES) * 512 + lane * 16;

  v8f acc[MT];
  #pragma unroll
  for (int t = 0; t < MT; ++t)
    acc[t] = v8f{0.f, 0.f, 0.f, 0.f, 0.f, 0.f, 0.f, 0.f};

  for (int kt = 0; kt < KTILES; ++kt) {
    union { uint4 q[2]; unsigned short us[16]; v16bf v; } bf;
    const int kb = kt * 32 + 16 * g;          // this half's K base
    if (SCALARB) {
      // conv1: kk == tap (CI=1); gather 16 f32 taps from x, convert
      #pragma unroll
      for (int e = 0; e < 16; ++e) {
        const int kk = kb + e;
        const int kh = (kk & 15) >> 2, kw = kk & 3;
        const int ih = ohb + kh, iw = owb + kw;
        const bool ok = (kk < 16) & ((unsigned)ih < (unsigned)H) &
                        ((unsigned)iw < (unsigned)W);
        float v = inf[((size_t)b * H + iclamp(ih, 0, H - 1)) * W +
                      iclamp(iw, 0, W - 1)];
        bf.us[e] = f2bf(ok ? v : 0.0f);
      }
    } else {
      // vector path: one tap, 16 consecutive channels -> 2x b128
      const int tap = kb / CI;                // const divisor, block-uniform
      const int ci0 = kb % CI;
      int ih, iw;
      if (MODE == 0) {
        const int kh = tap / KW, kw = tap % KW;
        ih = ohb + kh; iw = owb + kw;
      } else {
        ih = ohb - (tap >> 1); iw = owb - (tap & 1);
      }
      const bool ok = NOCHK | (((unsigned)ih < (unsigned)H) &
                               ((unsigned)iw < (unsigned)W));
      const unsigned short* src =
          inb + ((size_t)(b * H + iclamp(ih, 0, H - 1)) * W +
                 iclamp(iw, 0, W - 1)) * CI + ci0;
      uint4 q0 = *(const uint4*)src;
      uint4 q1 = *(const uint4*)(src + 8);
      if (!NOCHK) {
        q0.x = ok ? q0.x : 0u; q0.y = ok ? q0.y : 0u;
        q0.z = ok ? q0.z : 0u; q0.w = ok ? q0.w : 0u;
        q1.x = ok ? q1.x : 0u; q1.y = ok ? q1.y : 0u;
        q1.z = ok ? q1.z : 0u; q1.w = ok ? q1.w : 0u;
      }
      bf.q[0] = q0; bf.q[1] = q1;
    }
    // ---- MT WMMAs share this B fragment; A-fragments come from LDS ----
    #pragma unroll
    for (int t = 0; t < MT; ++t) {
      v16bf afrag = *(const v16bf*)(ap + (size_t)t * KTILES * 512);
      acc[t] = __builtin_amdgcn_wmma_f32_16x16x32_bf16(
          false, afrag, false, bf.v, (short)0, acc[t], false, false);
    }
    ap += 512;
  }

  if (EP == 2) {
    // d(m) = ||w_m||^2 - 2*(w_m . z_col); argmin over this lane's MT*8 rows
    float best = 3.4e38f; int bm = 0;
    #pragma unroll
    for (int t = 0; t < MT; ++t) {
      #pragma unroll
      for (int r = 0; r < 8; ++r) {
        const int m = (mgrp * MT + t) * 16 + 8 * g + r;
        const float d = cnorm[m] - 2.0f * acc[t][r];
        if (d < best) { best = d; bm = m; }
      }
    }
    unsigned fb  = __float_as_uint(best);
    unsigned ord = (fb & 0x80000000u) ? ~fb : (fb | 0x80000000u);
    unsigned long long key = ((unsigned long long)ord << 32) | (unsigned)bm;
    atomicMin(&vqmin[col], key);
  } else {
    // bf16 NHWC store: lane's 8 rows are 8 consecutive channels -> 1x b128/t
    unsigned short* ob = (unsigned short*)outv;
    int sh, sw;
    if (MODE == 0) { sh = oh; sw = ow; }
    else           { sh = 2 * oh + AA; sw = 2 * ow + BB; }
    const size_t nbase = ((size_t)(b * OH + sh) * OW + sw) * CO;
    #pragma unroll
    for (int t = 0; t < MT; ++t) {
      const int m0 = (mgrp * MT + t) * 16 + 8 * g;
      const float4 b0 = *(const float4*)(bias + m0);
      const float4 b1 = *(const float4*)(bias + m0 + 4);
      float vv[8];
      vv[0] = acc[t][0] + b0.x; vv[1] = acc[t][1] + b0.y;
      vv[2] = acc[t][2] + b0.z; vv[3] = acc[t][3] + b0.w;
      vv[4] = acc[t][4] + b1.x; vv[5] = acc[t][5] + b1.y;
      vv[6] = acc[t][6] + b1.z; vv[7] = acc[t][7] + b1.w;
      if (EP == 0) {
        #pragma unroll
        for (int r = 0; r < 8; ++r) vv[r] = fmaxf(vv[r], 0.0f);
      }
      uint4 q;
      q.x = (unsigned)f2bf(vv[0]) | ((unsigned)f2bf(vv[1]) << 16);
      q.y = (unsigned)f2bf(vv[2]) | ((unsigned)f2bf(vv[3]) << 16);
      q.z = (unsigned)f2bf(vv[4]) | ((unsigned)f2bf(vv[5]) << 16);
      q.w = (unsigned)f2bf(vv[6]) | ((unsigned)f2bf(vv[7]) << 16);
      *(uint4*)(ob + nbase + m0) = q;
    }
  }
}

// ---------------------------------------------------------------------------
// Small helper kernels
// ---------------------------------------------------------------------------
__global__ void cb_norms(const float* __restrict__ cb, float* __restrict__ cnorm) {
  int m = blockIdx.x * 256 + threadIdx.x;
  if (m >= 512) return;
  float s = 0.0f;
  for (int c = 0; c < 64; ++c) { float v = cb[m * 64 + c]; s += v * v; }
  cnorm[m] = s;
}

__global__ void vq_init(unsigned long long* v, int n) {
  int i = blockIdx.x * 256 + threadIdx.x;
  if (i < n) v[i] = 0xFFFFFFFFFFFFFFFFull;
}

// Extract argmin; emit indices (f32), z_q (f32 NCHW to d_out) and bf16 NHWC
// z_q for the decoder.
__global__ void vq_finalize(const unsigned long long* __restrict__ vqmin,
                            const float* __restrict__ cb,
                            unsigned short* __restrict__ zq_bf,
                            float* __restrict__ out_zq,
                            float* __restrict__ out_idx) {
  int n = blockIdx.x * 256 + threadIdx.x;
  if (n >= 65536) return;
  unsigned idx = (unsigned)(vqmin[n] & 0xFFFFFFFFull);
  out_idx[n] = (float)idx;
  int b = n >> 10, hw = n & 1023;
  for (int c = 0; c < 64; ++c) {
    float v = cb[(size_t)idx * 64 + c];
    out_zq[(((size_t)b * 64 + c) << 10) + hw] = v;     // f32 NCHW output
    zq_bf[(size_t)n * 64 + c] = f2bf(v);               // bf16 NHWC for dec1
  }
}

// dec3: convT (32->1 ch, k=4, s=2, p=1) + sigmoid, bf16 NHWC input.
__global__ void dec3_sig(const unsigned short* __restrict__ in,
                         const float* __restrict__ w,
                         const float* __restrict__ b3, float* __restrict__ out) {
  int i = blockIdx.x * 256 + threadIdx.x;
  if (i >= 64 * 128 * 128) return;
  int b  = i >> 14, rem = i & 16383;
  int oh = rem >> 7, ow = rem & 127;
  int a = oh & 1, c2 = ow & 1;
  int ohh = oh >> 1, oww = ow >> 1;
  int ohb = a  ? ohh + 1 : ohh;
  int owb = c2 ? oww + 1 : oww;
  float s = b3[0];
  #pragma unroll
  for (int ti = 0; ti < 2; ++ti) {
    int ih = ohb - ti;
    if ((unsigned)ih >= 64u) continue;
    int kh = a ? 2 * ti : 1 + 2 * ti;
    #pragma unroll
    for (int tj = 0; tj < 2; ++tj) {
      int iw = owb - tj;
      if ((unsigned)iw >= 64u) continue;
      int kw = c2 ? 2 * tj : 1 + 2 * tj;
      const unsigned short* src = in + ((size_t)(b * 64 + ih) * 64 + iw) * 32;
      for (int ci = 0; ci < 32; ++ci)
        s += bf2f(src[ci]) * w[(ci * 4 + kh) * 4 + kw];
    }
  }
  out[i] = 1.0f / (1.0f + __expf(-s));
}

// ---------------------------------------------------------------------------
// Host side
// ---------------------------------------------------------------------------
static inline int cdiv(int a, int b) { return (a + b - 1) / b; }

extern "C" void kernel_launch(void* const* d_in, const int* in_sizes, int n_in,
                              void* d_out, int out_size, void* d_ws, size_t ws_size,
                              hipStream_t stream) {
  const float* x      = (const float*)d_in[0];
  const float* enc_w1 = (const float*)d_in[1];
  const float* enc_b1 = (const float*)d_in[2];
  const float* enc_w2 = (const float*)d_in[3];
  const float* enc_b2 = (const float*)d_in[4];
  const float* enc_w3 = (const float*)d_in[5];
  const float* enc_b3 = (const float*)d_in[6];
  const float* cb     = (const float*)d_in[7];
  const float* dec_w1 = (const float*)d_in[8];
  const float* dec_b1 = (const float*)d_in[9];
  const float* dec_w2 = (const float*)d_in[10];
  const float* dec_b2 = (const float*)d_in[11];
  const float* dec_w3 = (const float*)d_in[12];
  const float* dec_b3 = (const float*)d_in[13];

  char* ws = (char*)d_ws;
  // bf16 NHWC activation buffers
  unsigned short* buf0 = (unsigned short*)(ws);             // 64x64x64x32 (h1 / d2out)
  unsigned short* buf1 = (unsigned short*)(ws + 16777216);  // 64x32x32x64 (h2 / d1out)
  unsigned short* buf2 = (unsigned short*)(ws + 25165824);  // z_e NHWC
  unsigned short* buf3 = (unsigned short*)(ws + 33554432);  // z_q NHWC
  unsigned long long* vqmin = (unsigned long long*)(ws + 41943040);   // 65536 u64
  float* cnorm = (float*)(ws + 42467328);                   // 512 f32
  unsigned short* ap = (unsigned short*)(ws + 42469376);
  unsigned short* apc1 = ap;                        // 32 x 32
  unsigned short* apc2 = ap + 1024;                 // 64 x 512
  unsigned short* apc3 = ap + 33792;                // 64 x 576
  unsigned short* apd1 = ap + 70656;                // 64 x 576
  unsigned short* apd2 = ap + 107520;               // 4 phases x (32 x 256)
  unsigned short* apvq = ap + 140288;               // 512 x 64

  float* out_xr  = (float*)d_out;                   // 1048576
  float* out_zq  = out_xr + 1048576;                // 4194304
  float* out_idx = out_zq + 4194304;                // 65536

  // ---- weight packing (tap-major K) ----
  { PackP p{}; p.w=enc_w1; p.dst=apc1; p.mode=0; p.Co=32; p.Ci=1;  p.KH=4; p.KW=4;
    p.Kpad=32;  p.Kreal=16;  p.Ktiles=1;
    pack_a<<<cdiv(32*32,256),256,0,stream>>>(p); }
  { PackP p{}; p.w=enc_w2; p.dst=apc2; p.mode=0; p.Co=64; p.Ci=32; p.KH=4; p.KW=4;
    p.Kpad=512; p.Kreal=512; p.Ktiles=16;
    pack_a<<<cdiv(64*512,256),256,0,stream>>>(p); }
  { PackP p{}; p.w=enc_w3; p.dst=apc3; p.mode=0; p.Co=64; p.Ci=64; p.KH=3; p.KW=3;
    p.Kpad=576; p.Kreal=576; p.Ktiles=18;
    pack_a<<<cdiv(64*576,256),256,0,stream>>>(p); }
  { PackP p{}; p.w=dec_w1; p.dst=apd1; p.mode=1; p.Co=64; p.Ci=64; p.KH=3; p.KW=3;
    p.Kpad=576; p.Kreal=576; p.Ktiles=18;
    pack_a<<<cdiv(64*576,256),256,0,stream>>>(p); }
  for (int ph = 0; ph < 4; ++ph) {
    PackP p{}; p.w=dec_w2; p.dst=apd2 + ph*8192; p.mode=2; p.Co=32; p.Ci=64;
    p.KH=4; p.KW=4; p.Kpad=256; p.Kreal=256; p.Ktiles=8;
    p.a = ph >> 1; p.b2 = ph & 1;
    pack_a<<<cdiv(32*256,256),256,0,stream>>>(p);
  }
  { PackP p{}; p.w=cb; p.dst=apvq; p.mode=3; p.Co=512; p.Ci=64; p.KH=1; p.KW=1;
    p.Kpad=64; p.Kreal=64; p.Ktiles=2;
    pack_a<<<cdiv(512*64,256),256,0,stream>>>(p); }

  cb_norms<<<2,256,0,stream>>>(cb, cnorm);
  vq_init<<<256,256,0,stream>>>(vqmin, 65536);

  // conv1: x(64,1,128,128 f32) -> h1 bf16 NHWC (64,64,64,32), relu.  MT=2.
  wmma_gemm<0,0,2,1, 1,128,128, 12,6, 4,4,2,1, 1, 64,64,32, 0,0>
      <<<4096,128,0,stream>>>(apc1, x, enc_b1, buf0, nullptr, nullptr);
  // conv2: h1 -> h2 bf16 NHWC (64,32,32,64), relu.  MT=4.
  wmma_gemm<0,0,4,1, 32,64,64, 10,5, 4,4,2,1, 16, 32,32,64, 0,0>
      <<<1024,128,0,stream>>>(apc2, buf0, enc_b2, buf1, nullptr, nullptr);
  // conv3: h2 -> z_e bf16 NHWC (64,32,32,64).  MT=4.
  wmma_gemm<0,1,4,1, 64,32,32, 10,5, 3,3,1,1, 18, 32,32,64, 0,0>
      <<<1024,128,0,stream>>>(apc3, buf1, enc_b3, buf2, nullptr, nullptr);
  // VQ: codebook(512x64) x z^T with fused argmin.  MT=8, GROUPS=4
  // (block stages the whole 64KB codebook once; its 4 waves cover all rows).
  wmma_gemm<0,2,8,4, 64,32,32, 10,5, 1,1,1,0, 2, 32,32,512, 0,0>
      <<<4096,128,0,stream>>>(apvq, buf2, nullptr, nullptr, vqmin, cnorm);
  vq_finalize<<<256,256,0,stream>>>(vqmin, cb, buf3, out_zq, out_idx);
  // dec1: z_q -> bf16 NHWC (64,32,32,64), relu (convT s=1 as flipped conv).
  wmma_gemm<0,0,4,1, 64,32,32, 10,5, 3,3,1,1, 18, 32,32,64, 0,0>
      <<<1024,128,0,stream>>>(apd1, buf3, dec_b1, buf1, nullptr, nullptr);
  // dec2: convT s=2 -> bf16 NHWC (64,64,64,32), relu; 4 parity phases.  MT=2.
  wmma_gemm<1,0,2,1, 64,32,32, 10,5, 4,4,2,1, 8, 64,64,32, 0,0>
      <<<1024,128,0,stream>>>(apd2 + 0*8192, buf1, dec_b2, buf0, nullptr, nullptr);
  wmma_gemm<1,0,2,1, 64,32,32, 10,5, 4,4,2,1, 8, 64,64,32, 0,1>
      <<<1024,128,0,stream>>>(apd2 + 1*8192, buf1, dec_b2, buf0, nullptr, nullptr);
  wmma_gemm<1,0,2,1, 64,32,32, 10,5, 4,4,2,1, 8, 64,64,32, 1,0>
      <<<1024,128,0,stream>>>(apd2 + 2*8192, buf1, dec_b2, buf0, nullptr, nullptr);
  wmma_gemm<1,0,2,1, 64,32,32, 10,5, 4,4,2,1, 8, 64,64,32, 1,1>
      <<<1024,128,0,stream>>>(apd2 + 3*8192, buf1, dec_b2, buf0, nullptr, nullptr);
  // dec3: convT s=2 (32->1) + sigmoid -> x_recon f32 (64,1,128,128)
  dec3_sig<<<cdiv(64*128*128,256),256,0,stream>>>(buf0, dec_w3, dec_b3, out_xr);
}
